// SignConnector_2817498546768
// MI455X (gfx1250) — matrix operand order repaired and checked by reading, not compiled, not run on hardware
//
#include <hip/hip_runtime.h>
#include <hip/hip_bf16.h>

// ---------------------------------------------------------------------------
// Problem constants (from the reference)
// ---------------------------------------------------------------------------
#define BB    4096      // batch
#define NN    46        // nodes
#define NP    48        // nodes padded (M dim)
#define NK    64        // nodes padded (K dim for A-multiplies)
#define CC    14        // input features
#define CKP   32        // input features padded to one WMMA K-step
#define HH    256       // hidden
#define EE    45        // edges
#define FC1N  128
#define FC2N  64
#define FLATK (NN * HH) // 11776

typedef _Float16 v16h __attribute__((ext_vector_type(16)));
typedef _Float16 v8h  __attribute__((ext_vector_type(8)));
typedef float    v8f  __attribute__((ext_vector_type(8)));
typedef int      v4i  __attribute__((ext_vector_type(4)));

union F16x16 { v16h v; _Float16 e[16]; };
union F16x8  { v8h  v; _Float16 e[8];  };
union F32x8  { v8f  v; float    e[8];  };

#define AS1 __attribute__((address_space(1)))
#define AS3 __attribute__((address_space(3)))

#if __has_builtin(__builtin_amdgcn_global_load_async_to_lds_b128)
#define USE_ASYNC_LDS 1
#else
#define USE_ASYNC_LDS 0
#endif

// ---------------------------------------------------------------------------
// WMMA helpers (CDNA5 wave32, V_WMMA_F32_16X16X32_F16)
// ---------------------------------------------------------------------------
__device__ __forceinline__ v8f wmma_f16(v16h a, v16h b, v8f c) {
  // 8 args: (neg_a, A, neg_b, B, c_mod, C, reuse_a, reuse_b)
  return __builtin_amdgcn_wmma_f32_16x16x32_f16(false, a, false, b,
                                                (short)0, c, false, false);
}

// A-fragment: 16x32 f16, row-major source (lda in elements, K contiguous).
// K order used here (contiguous, not ISA table order; matches B): lane -> row m = lane&15; half = lane>>4 selects K groups
//   halfs i=0..15 : K = half*16 + i     (contiguous 16 halfs)
__device__ __forceinline__ v16h load_frag_a(const _Float16* A, int lda,
                                            int m_base, int k_base) {
  int lane = threadIdx.x & 31;
  int m = lane & 15, h = lane >> 4;
  const _Float16* p = A + (size_t)(m_base + m) * lda + k_base + h * 16;
  F16x8 lo, hi;
  lo.v = *(const v8h*)(p);
  hi.v = *(const v8h*)(p + 8);
  F16x16 r;
#pragma unroll
  for (int i = 0; i < 8; ++i) { r.e[i] = lo.e[i]; r.e[8 + i] = hi.e[i]; }
  return r.v;
}

// B-fragment: 32x16 f16, from TRANSPOSED source Bt[n][k] (K contiguous).
// Same contiguous K order as A (not ISA table order): lane -> col n = lane&15; halfs i=0..15 -> K = (lane>>4)*16 + i
__device__ __forceinline__ v16h load_frag_b(const _Float16* Bt, int ldb,
                                            int n_base, int k_base) {
  int lane = threadIdx.x & 31;
  int n = lane & 15, h = lane >> 4;
  return *(const v16h*)(Bt + (size_t)(n_base + n) * ldb + k_base + h * 16);
}

// D store, transposed (column-major) as f16: Dt[n][m], m contiguous.
// D layout: vgpr r, lane -> (m = r + 8*(lane>>4), n = lane&15).
// Optionally zero-fills m in [48,64) so the next A-multiply's K-pad is clean.
__device__ __forceinline__ void store_d_ct(_Float16* Dt, int ldd,
                                           int n_base, int m_base,
                                           v8f c, bool pad48) {
  int lane = threadIdx.x & 31;
  int n = lane & 15, h = lane >> 4;
  F32x8 cc; cc.v = c;
  F16x8 o;
#pragma unroll
  for (int r = 0; r < 8; ++r) o.e[r] = (_Float16)cc.e[r];
  _Float16* row = Dt + (size_t)(n_base + n) * ldd;
  *(v8h*)(row + m_base + h * 8) = o.v;
  if (pad48) {
    F16x8 z = {};
    *(v8h*)(row + 48 + h * 8) = z.v;
  }
}

// D store, row-major f16 with bias + relu, rows clipped to mClip.
__device__ __forceinline__ void store_d_rm_bias_relu(_Float16* D, int ldd,
                                                     int m_base, int n_base,
                                                     v8f c, const float* bias,
                                                     int mClip) {
  int lane = threadIdx.x & 31;
  int n = lane & 15, h = lane >> 4;
  float bv = bias[n_base + n];
  F32x8 cc; cc.v = c;
#pragma unroll
  for (int r = 0; r < 8; ++r) {
    int m = m_base + h * 8 + r;
    if (m < mClip) {
      float v = cc.e[r] + bv;
      v = v > 0.0f ? v : 0.0f;
      D[(size_t)m * ldd + n_base + n] = (_Float16)v;
    }
  }
}

// ---------------------------------------------------------------------------
// Async global->LDS staging (gfx1250 ASYNCcnt path, with safe fallback)
// ---------------------------------------------------------------------------
__device__ __forceinline__ void stage_b128(const _Float16* g, _Float16* l) {
#if USE_ASYNC_LDS
  // Prototype (from clang): (int4 addrspace(1)*, int4 addrspace(3)*, Ii, Ii)
  _Float16* gnc = const_cast<_Float16*>(g);
  __builtin_amdgcn_global_load_async_to_lds_b128(
      (AS1 v4i*)gnc, (AS3 v4i*)l, 0, 0);
#else
  *(v8h*)l = *(const v8h*)g;  // global_load_b128 + ds_store_b128
#endif
}

__device__ __forceinline__ void async_wait0() {
#if USE_ASYNC_LDS
#if __has_builtin(__builtin_amdgcn_s_wait_asynccnt)
  __builtin_amdgcn_s_wait_asynccnt(0);
#else
  asm volatile("s_wait_asynccnt 0x0" ::: "memory");
#endif
#endif
}

__device__ __forceinline__ void async_wait2() {
#if USE_ASYNC_LDS
#if __has_builtin(__builtin_amdgcn_s_wait_asynccnt)
  __builtin_amdgcn_s_wait_asynccnt(2);
#else
  asm volatile("s_wait_asynccnt 0x2" ::: "memory");
#endif
#endif
}

// ---------------------------------------------------------------------------
// Prep kernels
// ---------------------------------------------------------------------------

// Build normalized adjacency (48x64, f16, zero-padded) in LDS then spill.
__global__ void adj_kernel(const long long* ei, const float* ew, _Float16* adjh) {
  __shared__ float deg[NP];
  __shared__ float Ash[NP * NK];
  int t = threadIdx.x; // 64 threads
  for (int i = t; i < NP; i += 64) deg[i] = 0.0f;
  for (int i = t; i < NP * NK; i += 64) Ash[i] = 0.0f;
  __syncthreads();
  for (int e = t; e < EE; e += 64) atomicAdd(&deg[(int)ei[EE + e]], ew[e]);
  for (int n = t; n < NN; n += 64) atomicAdd(&deg[n], 1.0f);
  __syncthreads();
  for (int n = t; n < NN; n += 64) deg[n] = deg[n] > 0.0f ? rsqrtf(deg[n]) : 0.0f;
  __syncthreads();
  for (int e = t; e < EE; e += 64) {
    int r = (int)ei[e], c = (int)ei[EE + e];
    atomicAdd(&Ash[c * NK + r], deg[r] * ew[e] * deg[c]); // A[cc][rr] += norm
  }
  for (int n = t; n < NN; n += 64) atomicAdd(&Ash[n * NK + n], deg[n] * deg[n]);
  __syncthreads();
  for (int i = t; i < NP * NK; i += 64) adjh[i] = (_Float16)Ash[i];
}

// Per-sample coordinate normalization + cast to f16, padded (48 x 32).
__global__ void coordnorm_kernel(const float* x, _Float16* xh) {
  int b = blockIdx.x;
  int t = threadIdx.x; // 64 threads
  __shared__ float sx[NP], sy[NP], sz[NP], sn[NP];
  __shared__ float red[4];
  const float* xb = x + (size_t)b * NN * CC;
  if (t < NN) {
    sx[t] = xb[t * CC + 0];
    sy[t] = xb[t * CC + 1];
    sz[t] = xb[t * CC + 2];
  }
  __syncthreads();
  if (t == 0) {
    float ax = 0, ay = 0, az = 0;
    for (int n = 0; n < NN; ++n) { ax += sx[n]; ay += sy[n]; az += sz[n]; }
    red[0] = ax / (float)NN; red[1] = ay / (float)NN; red[2] = az / (float)NN;
  }
  __syncthreads();
  float cx = red[0], cy = red[1], cz = red[2];
  if (t < NN) {
    float dx = sx[t] - cx, dy = sy[t] - cy, dz = sz[t] - cz;
    sn[t] = sqrtf(dx * dx + dy * dy + dz * dz);
  }
  __syncthreads();
  if (t == 0) {
    float s = 0;
    for (int n = 0; n < NN; ++n) s += sn[n];
    red[3] = s / (float)NN + 1e-6f;
  }
  __syncthreads();
  float inv = 1.0f / red[3];
  _Float16* ob = xh + (size_t)b * NP * CKP;
  for (int i = t; i < NP * CKP; i += 64) {
    int node = i >> 5, k = i & 31;
    float v = 0.0f;
    if (node < NN && k < CC) {
      if (k < 3) {
        float c0 = (k == 0) ? cx : ((k == 1) ? cy : cz);
        v = (xb[node * CC + k] - c0) * inv;
      } else {
        v = xb[node * CC + k];
      }
    }
    ob[i] = (_Float16)v;
  }
}

// Transpose + f32->f16 cast of weights: W (Kin x Nout) -> WT (Nout x Kpad).
__global__ void wtrans_kernel(const float* W, _Float16* WT,
                              int Kin, int Kpad, int Nout) {
  int total = Nout * Kpad;
  for (int i = blockIdx.x * blockDim.x + threadIdx.x; i < total;
       i += gridDim.x * blockDim.x) {
    int n = i / Kpad, k = i % Kpad;
    WT[i] = (k < Kin) ? (_Float16)W[(size_t)k * Nout + n] : (_Float16)0.0f;
  }
}

// ---------------------------------------------------------------------------
// WMMA GEMM kernels. One wave per 16x16 D tile, blockDim = (32, wavesPerBlk).
// ---------------------------------------------------------------------------

// D^T = (A @ B)^T, stored column-major f16 with node-pad zero-fill at mt==2.
// Used for:  T1 = Xn @ W1   and   T2 = H1 @ W2   (per batch, M=48, N=256)
__global__ void gemm_ct_kernel(const _Float16* A, size_t aStride, int lda,
                               const _Float16* Bt, int ldb, int K,
                               _Float16* Dt, size_t dStride, int ldd,
                               int nTiles) {
  int b = blockIdx.y;
  int tile = blockIdx.x * blockDim.y + threadIdx.y;
  int mt = tile / nTiles, nt = tile % nTiles;
  const _Float16* Ab = A + aStride * (size_t)b;
  v8f c = {};
  for (int k = 0; k < K; k += 32) {
    v16h af = load_frag_a(Ab, lda, mt * 16, k);
    v16h bf = load_frag_b(Bt, ldb, nt * 16, k);
    c = wmma_f16(af, bf, c);
  }
  store_d_ct(Dt + dStride * (size_t)b, ldd, nt * 16, mt * 16, c, mt == 2);
}

// D = relu(A @ B + bias), row-major f16, rows clipped to mClip.
// Used for: H = relu(A_adj @ T + b)  (per batch)  and  FC2.
__global__ void gemm_rm_kernel(const _Float16* A, size_t aStride, int lda,
                               const _Float16* Bt, size_t bStride, int ldb,
                               int K, const float* bias,
                               _Float16* D, size_t dStride, int ldd,
                               int nTiles, int mClip) {
  int b = blockIdx.y;
  int tile = blockIdx.x * blockDim.y + threadIdx.y;
  int mt = tile / nTiles, nt = tile % nTiles;
  const _Float16* Ab = A + aStride * (size_t)b;
  const _Float16* Bb = Bt + bStride * (size_t)b;
  v8f c = {};
  for (int k = 0; k < K; k += 32) {
    v16h af = load_frag_a(Ab, lda, mt * 16, k);
    v16h bf = load_frag_b(Bb, ldb, nt * 16, k);
    c = wmma_f16(af, bf, c);
  }
  store_d_rm_bias_relu(D + dStride * (size_t)b, ldd, mt * 16, nt * 16, c,
                       bias, mClip);
}

// ---------------------------------------------------------------------------
// FC1 with LDS-staged A-tile (double-buffered async global->LDS DMA).
// All 8 waves of a block share one M-tile; A is fetched once per block.
// K = 11776 = 46 stages x 256 halfs.
// ---------------------------------------------------------------------------
#define FC1_KB   256            // K-halfs per stage
#define FC1_LDA  264            // padded LDS row (avoids all-lanes-same-bank)
#define FC1_S    (FLATK / FC1_KB)  // 46 stages

__global__ void __launch_bounds__(256) fc1_kernel(const _Float16* Ag,
                                                  const _Float16* Bt,
                                                  const float* bias,
                                                  _Float16* D) {
  __shared__ _Float16 As[2][16 * FC1_LDA];  // 2 x 8.25 KB
  int mt = blockIdx.x;          // 0..255 (M tile)
  int nt = threadIdx.y;         // 0..7   (N tile)
  int tid = threadIdx.y * 32 + threadIdx.x;
  int row = tid >> 4;           // 0..15
  int seg = tid & 15;           // 0..15 (two b128 per thread per stage)
  const _Float16* gRow = Ag + (size_t)(mt * 16 + row) * FLATK;

  v8f c = {};

  // stage 0 issue
  {
    const _Float16* g = gRow + seg * 8;
    _Float16* l = &As[0][row * FC1_LDA + seg * 8];
    stage_b128(g, l);
    stage_b128(g + 128, l + 128);
  }

  for (int s = 0; s < FC1_S; ++s) {
    if (s + 1 < FC1_S) {
      const _Float16* g = gRow + (s + 1) * FC1_KB + seg * 8;
      _Float16* l = &As[(s + 1) & 1][row * FC1_LDA + seg * 8];
      stage_b128(g, l);
      stage_b128(g + 128, l + 128);
      async_wait2();   // stage s complete, stage s+1 still in flight
    } else {
      async_wait0();
    }
    __syncthreads();

    const _Float16* Asb = &As[s & 1][0];
    int kg = s * FC1_KB;
#pragma unroll
    for (int kk = 0; kk < FC1_KB; kk += 32) {
      v16h af = load_frag_a(Asb, FC1_LDA, 0, kk);
      v16h bf = load_frag_b(Bt, FLATK, nt * 16, kg + kk);
      c = wmma_f16(af, bf, c);
    }
    __syncthreads();  // all reads of buffer s&1 done before stage s+2 rewrites
  }

  store_d_rm_bias_relu(D, FC1N, mt * 16, nt * 16, c, bias, BB);
}

// Final 64->1 projection in f32.
__global__ void fc3_kernel(const _Float16* o2, const float* fW3,
                           const float* fb3, float* out) {
  int i = blockIdx.x * blockDim.x + threadIdx.x;
  if (i >= BB) return;
  float acc = fb3[0];
  const _Float16* p = o2 + (size_t)i * FC2N;
#pragma unroll 8
  for (int k = 0; k < FC2N; ++k) acc += (float)p[k] * fW3[k];
  out[i] = acc;
}

// ---------------------------------------------------------------------------
// Host launcher
// ---------------------------------------------------------------------------
extern "C" void kernel_launch(void* const* d_in, const int* in_sizes, int n_in,
                              void* d_out, int out_size, void* d_ws,
                              size_t ws_size, hipStream_t stream) {
  const float*     x   = (const float*)d_in[0];
  const long long* ei  = (const long long*)d_in[1]; // int64 edge_index (2,45)
  const float*     ew  = (const float*)d_in[2];
  const float*     W1  = (const float*)d_in[3];
  const float*     b1  = (const float*)d_in[4];
  const float*     W2  = (const float*)d_in[5];
  const float*     b2  = (const float*)d_in[6];
  const float*     fW1 = (const float*)d_in[7];
  const float*     fb1 = (const float*)d_in[8];
  const float*     fW2 = (const float*)d_in[9];
  const float*     fb2 = (const float*)d_in[10];
  const float*     fW3 = (const float*)d_in[11];
  const float*     fb3 = (const float*)d_in[12];
  float* out = (float*)d_out;

  // Workspace carve-up (all regions 64B aligned; sizes in f16 elements).
  _Float16* ws = (_Float16*)d_ws;
  size_t off = 0;
  auto alloc = [&](size_t halfs) {
    _Float16* p = ws + off;
    off += (halfs + 31) & ~(size_t)31;
    return p;
  };
  _Float16* xh   = alloc((size_t)BB * NP * CKP);  // padded f16 inputs
  _Float16* Tt   = alloc((size_t)BB * HH * NK);   // T1^T, then reused as T2^T
  _Float16* Hbuf = alloc((size_t)BB * NP * HH);   // H1, then reused as H2 (compact)
  _Float16* w1t  = alloc((size_t)HH * CKP);
  _Float16* w2t  = alloc((size_t)HH * HH);
  _Float16* fw1t = alloc((size_t)FC1N * FLATK);
  _Float16* fw2t = alloc((size_t)FC2N * FC1N);
  _Float16* o1   = alloc((size_t)BB * FC1N);
  _Float16* o2   = alloc((size_t)BB * FC2N);
  _Float16* adjh = alloc((size_t)NP * NK);

  dim3 wgemm(32, 8);

  // ---- prep ----
  adj_kernel<<<1, 64, 0, stream>>>(ei, ew, adjh);
  coordnorm_kernel<<<BB, 64, 0, stream>>>(x, xh);
  wtrans_kernel<<<64, 256, 0, stream>>>(W1, w1t, CC, CKP, HH);
  wtrans_kernel<<<256, 256, 0, stream>>>(W2, w2t, HH, HH, HH);
  wtrans_kernel<<<1024, 256, 0, stream>>>(fW1, fw1t, FLATK, FLATK, FC1N);
  wtrans_kernel<<<32, 256, 0, stream>>>(fW2, fw2t, FC1N, FC1N, FC2N);

  // ---- layer 1: T1 = Xn @ W1  (per batch 48x32 @ 32x256), store T1^T ----
  gemm_ct_kernel<<<dim3(6, BB), wgemm, 0, stream>>>(
      xh, (size_t)NP * CKP, CKP, w1t, CKP, CKP,
      Tt, (size_t)HH * NK, NK, 16);

  // ---- H1 = relu(A @ T1 + b1)  (per batch 48x64 @ 64x256) ----
  gemm_rm_kernel<<<dim3(6, BB), wgemm, 0, stream>>>(
      adjh, 0, NK, Tt, (size_t)HH * NK, NK, NK, b1,
      Hbuf, (size_t)NP * HH, HH, 16, NP);

  // ---- T2 = H1 @ W2  (per batch 48x256 @ 256x256), store T2^T (reuse Tt) ----
  gemm_ct_kernel<<<dim3(6, BB), wgemm, 0, stream>>>(
      Hbuf, (size_t)NP * HH, HH, w2t, HH, HH,
      Tt, (size_t)HH * NK, NK, 16);

  // ---- H2 = relu(A @ T2 + b2), stored COMPACT (B x 11776) for the flatten ----
  gemm_rm_kernel<<<dim3(6, BB), wgemm, 0, stream>>>(
      adjh, 0, NK, Tt, (size_t)HH * NK, NK, NK, b2,
      Hbuf, (size_t)FLATK, HH, 16, NN);

  // ---- FC1: (4096 x 11776) @ (11776 x 128), relu; LDS-staged A tile ----
  fc1_kernel<<<256, wgemm, 0, stream>>>(Hbuf, fw1t, fb1, o1);

  // ---- FC2: (4096 x 128) @ (128 x 64), relu ----
  gemm_rm_kernel<<<dim3(128, 1), wgemm, 0, stream>>>(
      o1, 0, FC1N, fw2t, 0, FC1N, FC1N, fb2,
      o2, 0, FC2N, FC2N / 16, BB);

  // ---- FC3: (4096 x 64) @ (64 x 1) + fb3, f32 out ----
  fc3_kernel<<<16, 256, 0, stream>>>(o2, fW3, fb3, out);
}
